// Block_28346784154190
// MI455X (gfx1250) — matrix operand order repaired and checked by reading, not compiled, not run on hardware
//
#include <hip/hip_runtime.h>
#include <hip/hip_bf16.h>
#include <math.h>
#include <stdint.h>

typedef __bf16 bf16;
typedef __attribute__((ext_vector_type(16))) __bf16 v16bf;
typedef __attribute__((ext_vector_type(8)))  float  v8f;
typedef __attribute__((ext_vector_type(4)))  unsigned v4u;
typedef int v4i_g __attribute__((vector_size(16)));   // matches builtin pointee type

#define DIM      1024
#define HEADS    16
#define DH       64
#define TQ       2048
#define NB       2
#define BT       (NB * TQ)      /* 4096 rows */
#define PROJ     2728
#define PROJ_PAD 2816           /* pad to multiple of 128 */
#define EPSF     1e-5f

// ---------------- CDNA5 async global->LDS DMA (ASYNCcnt path) ----------------
#if defined(__gfx1250__) && __has_builtin(__builtin_amdgcn_global_load_async_to_lds_b128)
#define HAS_ASYNC_LDS 1
static __device__ inline void async_cp16(const bf16* g, bf16* l) {
    // AS1 VA == flat VA; AS3 offset == low 32 bits of flat LDS address (ISA 10.2)
    __builtin_amdgcn_global_load_async_to_lds_b128(
        (__attribute__((address_space(1))) v4i_g*)(unsigned long long)(uintptr_t)g,
        (__attribute__((address_space(3))) v4i_g*)(unsigned)(uintptr_t)l,
        0, 0);
}
static __device__ inline void wait_async(int n) {
#if __has_builtin(__builtin_amdgcn_s_wait_asynccnt)
    if (n == 0) __builtin_amdgcn_s_wait_asynccnt(0);
    else        __builtin_amdgcn_s_wait_asynccnt(4);
#else
    if (n == 0) asm volatile("s_wait_asynccnt 0x0" ::: "memory");
    else        asm volatile("s_wait_asynccnt 0x4" ::: "memory");
#endif
}
#else
#define HAS_ASYNC_LDS 0
static __device__ inline void wait_async(int) {}
#endif

// copy one 16-byte chunk global->LDS (async if available)
static __device__ inline void cp16(const bf16* g, bf16* l) {
#if HAS_ASYNC_LDS
    async_cp16(g, l);
#else
    *(v4u*)l = *(const v4u*)g;
#endif
}

// ---- fragment loader: two 16-byte LDS reads -> one 16xbf16 WMMA operand ----
static __device__ inline v16bf frag_ld(const bf16* p0, const bf16* p1) {
    union { v4u q[2]; v16bf v; } u;
    u.q[0] = *(const v4u*)p0;
    u.q[1] = *(const v4u*)p1;
    return u.v;
}
static __device__ inline v8f v8f_zero() {
    v8f v;
#pragma unroll
    for (int i = 0; i < 8; i++) v[i] = 0.0f;
    return v;
}

// -------- transpose + cast f32 [K][N] -> bf16 [Npad][Kpad], zero-padded --------
__global__ __launch_bounds__(256)
void tcast(const float* __restrict__ src, bf16* __restrict__ dst,
           int K, int N, int Kpad, int Npad) {
    __shared__ float tl[32][33];
    const int nb = blockIdx.x * 32, kb = blockIdx.y * 32;
    const int tx = threadIdx.x, ty = threadIdx.y;
#pragma unroll
    for (int i = 0; i < 32; i += 8) {
        int k = kb + ty + i, n = nb + tx;
        tl[ty + i][tx] = (k < K && n < N) ? src[(long)k * N + n] : 0.0f;
    }
    __syncthreads();
#pragma unroll
    for (int i = 0; i < 32; i += 8) {
        int n = nb + ty + i, k = kb + tx;
        if (n < Npad && k < Kpad) dst[(long)n * Kpad + k] = (bf16)tl[tx][ty + i];
    }
}

// ---------------- RMSNorm (per-row, C=1024) + cast to bf16 ----------------
__global__ __launch_bounds__(256)
void rmsnorm_cast(const float* __restrict__ x, const float* __restrict__ g,
                  bf16* __restrict__ out) {
    __shared__ float red[256];
    const int row = blockIdx.x;
    const float4* x4 = (const float4*)(x + (long)row * DIM);
    float4 m = x4[threadIdx.x];
    float s = m.x * m.x + m.y * m.y + m.z * m.z + m.w * m.w;
    red[threadIdx.x] = s;
    __syncthreads();
    for (int off = 128; off > 0; off >>= 1) {
        if (threadIdx.x < off) red[threadIdx.x] += red[threadIdx.x + off];
        __syncthreads();
    }
    float rms = sqrtf(red[0] * (1.0f / DIM));
    float inv = 1.0f / (rms + EPSF);
    float4 gg = ((const float4*)g)[threadIdx.x];
    bf16* o = out + (long)row * DIM + threadIdx.x * 4;
    o[0] = (bf16)(m.x * inv * gg.x);
    o[1] = (bf16)(m.y * inv * gg.y);
    o[2] = (bf16)(m.z * inv * gg.z);
    o[3] = (bf16)(m.w * inv * gg.w);
}

// stage a 128x32 bf16 tile (row-major, ld = K) into LDS (stride 40), 256 threads
static __device__ inline void stage_tile(const bf16* __restrict__ gbase, long ld,
                                         long rowBase, int k0, bf16* lbuf, int t) {
    const int row = t >> 1;
    const int q0 = (t & 1) * 2;
#pragma unroll
    for (int c = 0; c < 2; c++) {
        int q = q0 + c; // 16B chunk index 0..3 within the 32-col row
        cp16(gbase + (rowBase + row) * ld + k0 + q * 8, &lbuf[row * 40 + q * 8]);
    }
}

// ------- bf16 WMMA GEMM: 128x128x32 block tile, 8 waves, double-buffered LDS ----
// B is pre-transposed: Bt[n][k], ld = K.
// EPI: 0=store f32, 1=f32 + residual, 2=store bf16, 3=bf16 silu(other)*acc
template <int EPI>
__global__ __launch_bounds__(256)
void gemm_bf16(const bf16* __restrict__ A, const bf16* __restrict__ Bt,
               float* __restrict__ outF, bf16* __restrict__ outB,
               const float* __restrict__ resid, const float* __restrict__ other,
               int M, int N, int K) {
    __shared__ __align__(16) bf16 sA[2][128 * 40];
    __shared__ __align__(16) bf16 sB[2][128 * 40];
    const int t = threadIdx.x;
    const int wid = t >> 5, lane = t & 31;
    const int lh = lane >> 4, l16 = lane & 15;
    const int wm = wid >> 1, wn = wid & 1;        // wave tile: 32 rows x 64 cols
    const long mBase = (long)blockIdx.y * 128;
    const long nBase = (long)blockIdx.x * 128;

    v8f acc[2][4];
#pragma unroll
    for (int i = 0; i < 2; i++)
#pragma unroll
        for (int j = 0; j < 4; j++) acc[i][j] = v8f_zero();

    const int nt = K / 32;
    stage_tile(A,  K, mBase, 0, sA[0], t);
    stage_tile(Bt, K, nBase, 0, sB[0], t);

    for (int it = 0; it < nt; ++it) {
        const int cur = it & 1;
        if (it + 1 < nt) {
            stage_tile(A,  K, mBase, (it + 1) * 32, sA[cur ^ 1], t);
            stage_tile(Bt, K, nBase, (it + 1) * 32, sB[cur ^ 1], t);
            wait_async(4);   // retire tile it (4 per-thread transfers per matrix pair)
        } else {
            wait_async(0);
        }
        __syncthreads();

        v16bf af[2], bfr[4];
#pragma unroll
        for (int am = 0; am < 2; am++) {
            const bf16* p = &sA[cur][(wm * 32 + am * 16 + l16) * 40 + lh * 8];
            af[am] = frag_ld(p, p + 16);
        }
#pragma unroll
        for (int bn = 0; bn < 4; bn++) {
            const bf16* p = &sB[cur][(wn * 64 + bn * 16 + l16) * 40 + lh * 8];
            bfr[bn] = frag_ld(p, p + 16);
        }
#pragma unroll
        for (int am = 0; am < 2; am++)
#pragma unroll
            for (int bn = 0; bn < 4; bn++)
                acc[am][bn] = __builtin_amdgcn_wmma_f32_16x16x32_bf16(
                    false, af[am], false, bfr[bn], (short)0, acc[am][bn], false, false);
        __syncthreads();   // allow next-next stage to overwrite this buffer
    }

#pragma unroll
    for (int am = 0; am < 2; am++)
#pragma unroll
        for (int bn = 0; bn < 4; bn++)
#pragma unroll
            for (int r = 0; r < 8; r++) {
                long m = mBase + wm * 32 + am * 16 + r + lh * 8;
                long n = nBase + wn * 64 + bn * 16 + l16;
                long idx = m * N + n;
                float v = acc[am][bn][r];
                if (EPI == 0) {
                    outF[idx] = v;
                } else if (EPI == 1) {
                    outF[idx] = v + resid[idx];
                } else if (EPI == 2) {
                    outB[idx] = (bf16)v;
                } else {
                    float o = other[idx];
                    float sg = o / (1.0f + __expf(-o));
                    outB[idx] = (bf16)(sg * v);
                }
            }
}

// ---------------- Flash attention: 64 queries/block, 4 waves, online softmax ----
__global__ __launch_bounds__(128)
void attn_kernel(const bf16* __restrict__ qkv, bf16* __restrict__ outc) {
    __shared__ __align__(16) bf16 sQ[64 * 72];
    __shared__ __align__(16) bf16 sK[64 * 72];
    __shared__ __align__(16) bf16 sVt[64 * 72];   // [d][k]
    __shared__ __align__(16) bf16 sP[64 * 72];
    __shared__ float sS[64 * 72];
    __shared__ float sM[64], sL[64], sAl[64];

    const int t = threadIdx.x;                    // 128 threads = 4 waves
    const int wid = t >> 5, lane = t & 31;
    const int lh = lane >> 4, l16 = lane & 15;
    const int qt = blockIdx.x, h = blockIdx.y, b = blockIdx.z;
    const long qbase = (long)qt * 64;
    const float slope = exp2f(-0.5f * (float)(h + 1)); // (2^(8/16))^-(h+1)
    const float invs = 1.0f / 32.0f;                   // 1/sqrt(C)
    const long rs = 3 * DIM;                           // qkv row stride

    { // stage Q tile (pure copy -> async path)
        int r = t >> 1;
#pragma unroll
        for (int c = 0; c < 4; c++) {
            int q = (t & 1) * 4 + c; // 8 chunks of 8 elems per 64-col row
            cp16(qkv + ((long)b * TQ + qbase + r) * rs + h * DH + q * 8,
                 &sQ[r * 72 + q * 8]);
        }
    }
    if (t < 64) { sM[t] = -INFINITY; sL[t] = 0.0f; }
    wait_async(0);
    __syncthreads();

    v16bf aq[2];
#pragma unroll
    for (int c = 0; c < 2; c++) {
        const bf16* p = &sQ[(wid * 16 + l16) * 72 + c * 32 + lh * 8];
        aq[c] = frag_ld(p, p + 16);
    }
    v8f o[4];
#pragma unroll
    for (int fn = 0; fn < 4; fn++) o[fn] = v8f_zero();

    for (int j = 0; j <= qt; j++) {
        __syncthreads();   // protect sK/sVt from previous PV read
        { // K tile (pure copy -> async path)
            int r = t >> 1;
#pragma unroll
            for (int c = 0; c < 4; c++) {
                int q = (t & 1) * 4 + c;
                cp16(qkv + ((long)b * TQ + (long)j * 64 + r) * rs + DIM + h * DH + q * 8,
                     &sK[r * 72 + q * 8]);
            }
        }
        { // V tile, transposed into [d][k] with k-pairs packed (register path)
            int kp = t & 31, dbase = (t >> 5) * 16;
            const bf16* g0 = qkv + ((long)b * TQ + (long)j * 64 + 2 * kp) * rs + 2 * DIM + h * DH + dbase;
            const bf16* g1 = g0 + rs;
            union { v4u q[2]; unsigned short s[16]; } r0, r1;
            r0.q[0] = ((const v4u*)g0)[0]; r0.q[1] = ((const v4u*)g0)[1];
            r1.q[0] = ((const v4u*)g1)[0]; r1.q[1] = ((const v4u*)g1)[1];
#pragma unroll
            for (int i = 0; i < 16; i++) {
                unsigned v = (unsigned)r0.s[i] | ((unsigned)r1.s[i] << 16);
                *(unsigned*)&sVt[(dbase + i) * 72 + 2 * kp] = v;
            }
        }
        wait_async(0);
        __syncthreads();

        // S = Q K^T  (wave: 16 query rows x 64 keys)
        v8f s4[4];
#pragma unroll
        for (int fn = 0; fn < 4; fn++) s4[fn] = v8f_zero();
#pragma unroll
        for (int c = 0; c < 2; c++)
#pragma unroll
            for (int fn = 0; fn < 4; fn++) {
                const bf16* p = &sK[(fn * 16 + l16) * 72 + c * 32 + lh * 8];
                v16bf bk = frag_ld(p, p + 16);
                s4[fn] = __builtin_amdgcn_wmma_f32_16x16x32_bf16(
                    false, aq[c], false, bk, (short)0, s4[fn], false, false);
            }
        // scale + ALiBi + causal mask -> sS
#pragma unroll
        for (int fn = 0; fn < 4; fn++)
#pragma unroll
            for (int r = 0; r < 8; r++) {
                int ml = r + lh * 8;
                long qpos = qbase + wid * 16 + ml;
                long kpos = (long)j * 64 + fn * 16 + l16;
                float v = s4[fn][r] * invs + slope * (float)(kpos - qpos);
                if (kpos > qpos) v = -INFINITY;
                sS[(wid * 16 + ml) * 72 + fn * 16 + l16] = v;
            }
        __syncthreads();

        // online softmax per row
        if (t < 64) {
            float mold = sM[t];
            const float* rowp = &sS[t * 72];
            float mj = -INFINITY;
            for (int kk = 0; kk < 64; kk++) mj = fmaxf(mj, rowp[kk]);
            float mnew = fmaxf(mold, mj);
            float alpha = (mold == -INFINITY) ? 0.0f : __expf(mold - mnew);
            float sum = 0.0f;
            bf16* pr = &sP[t * 72];
            for (int kk = 0; kk < 64; kk++) {
                float vv = rowp[kk];
                float p = (vv == -INFINITY) ? 0.0f : __expf(vv - mnew);
                pr[kk] = (bf16)p;
                sum += p;
            }
            sL[t] = alpha * sL[t] + sum;
            sM[t] = mnew;
            sAl[t] = alpha;
        }
        __syncthreads();

        // rescale O, then O += P V
#pragma unroll
        for (int r = 0; r < 8; r++) {
            float al = sAl[wid * 16 + r + lh * 8];
#pragma unroll
            for (int fn = 0; fn < 4; fn++) o[fn][r] *= al;
        }
        v16bf ap[2];
#pragma unroll
        for (int c = 0; c < 2; c++) {
            const bf16* p = &sP[(wid * 16 + l16) * 72 + c * 32 + lh * 8];
            ap[c] = frag_ld(p, p + 16);
        }
#pragma unroll
        for (int c = 0; c < 2; c++)
#pragma unroll
            for (int fn = 0; fn < 4; fn++) {
                const bf16* p = &sVt[(fn * 16 + l16) * 72 + c * 32 + lh * 8];
                v16bf bv = frag_ld(p, p + 16);
                o[fn] = __builtin_amdgcn_wmma_f32_16x16x32_bf16(
                    false, ap[c], false, bv, (short)0, o[fn], false, false);
            }
    }

    // normalize and write concat output [b,t, h*64+d] as bf16
#pragma unroll
    for (int r = 0; r < 8; r++) {
        int ml = r + lh * 8;
        float linv = 1.0f / sL[wid * 16 + ml];
        long qrow = (long)b * TQ + qbase + wid * 16 + ml;
#pragma unroll
        for (int fn = 0; fn < 4; fn++) {
            int d = fn * 16 + l16;
            outc[qrow * DIM + h * DH + d] = (bf16)(o[fn][r] * linv);
        }
    }
}

// ---------------- host orchestration ----------------
extern "C" void kernel_launch(void* const* d_in, const int* in_sizes, int n_in,
                              void* d_out, int out_size, void* d_ws, size_t ws_size,
                              hipStream_t stream) {
    const float* x    = (const float*)d_in[0];
    const float* g1   = (const float*)d_in[1];
    const float* wqkv = (const float*)d_in[2];
    const float* wo   = (const float*)d_in[3];
    const float* g2   = (const float*)d_in[4];
    const float* Wm   = (const float*)d_in[5];
    const float* Vm   = (const float*)d_in[6];
    const float* W2m  = (const float*)d_in[7];
    float* out = (float*)d_out;

    char* ws = (char*)d_ws;
    size_t off = 0;
    auto alloc = [&](size_t bytes) -> char* {
        char* p = ws + off;
        off += (bytes + 255) & ~(size_t)255;
        return p;
    };
    bf16*  hb     = (bf16*)alloc((size_t)BT * DIM * 2);          // norm output (reused)
    bf16*  wqkvT  = (bf16*)alloc((size_t)3 * DIM * DIM * 2);     // [3C][C]
    bf16*  woT    = (bf16*)alloc((size_t)DIM * DIM * 2);         // [C][C]
    bf16*  WT     = (bf16*)alloc((size_t)PROJ_PAD * DIM * 2);    // [Ppad][C]
    bf16*  VT     = (bf16*)alloc((size_t)PROJ_PAD * DIM * 2);    // [Ppad][C]
    bf16*  W2T    = (bf16*)alloc((size_t)DIM * PROJ_PAD * 2);    // [C][Ppad]
    bf16*  qkv_b  = (bf16*)alloc((size_t)BT * 3 * DIM * 2);
    bf16*  attn_b = (bf16*)alloc((size_t)BT * DIM * 2);
    float* x1     = (float*)alloc((size_t)BT * DIM * 4);
    float* tmp1   = (float*)alloc((size_t)BT * PROJ_PAD * 4);
    bf16*  up_b   = (bf16*)alloc((size_t)BT * PROJ_PAD * 2);

    // transpose-cast all weights once: src f32 [K][N] -> bf16 [Npad][Kpad]
    dim3 tb(32, 8);
    tcast<<<dim3(3 * DIM / 32, DIM / 32), tb, 0, stream>>>(wqkv, wqkvT, DIM, 3 * DIM, DIM, 3 * DIM);
    tcast<<<dim3(DIM / 32, DIM / 32), tb, 0, stream>>>(wo, woT, DIM, DIM, DIM, DIM);
    tcast<<<dim3(PROJ_PAD / 32, DIM / 32), tb, 0, stream>>>(Wm, WT, DIM, PROJ, DIM, PROJ_PAD);
    tcast<<<dim3(PROJ_PAD / 32, DIM / 32), tb, 0, stream>>>(Vm, VT, DIM, PROJ, DIM, PROJ_PAD);
    tcast<<<dim3(DIM / 32, PROJ_PAD / 32), tb, 0, stream>>>(W2m, W2T, PROJ, DIM, PROJ_PAD, DIM);

    // attention sublayer
    rmsnorm_cast<<<BT, 256, 0, stream>>>(x, g1, hb);
    gemm_bf16<2><<<dim3(3 * DIM / 128, BT / 128), 256, 0, stream>>>(
        hb, wqkvT, nullptr, qkv_b, nullptr, nullptr, BT, 3 * DIM, DIM);
    attn_kernel<<<dim3(TQ / 64, HEADS, NB), 128, 0, stream>>>(qkv_b, attn_b);
    gemm_bf16<1><<<dim3(DIM / 128, BT / 128), 256, 0, stream>>>(
        attn_b, woT, x1, nullptr, x, nullptr, BT, DIM, DIM);

    // SwiGLU sublayer
    rmsnorm_cast<<<BT, 256, 0, stream>>>(x1, g2, hb);
    gemm_bf16<0><<<dim3(PROJ_PAD / 128, BT / 128), 256, 0, stream>>>(
        hb, WT, tmp1, nullptr, nullptr, nullptr, BT, PROJ_PAD, DIM);
    gemm_bf16<3><<<dim3(PROJ_PAD / 128, BT / 128), 256, 0, stream>>>(
        hb, VT, nullptr, up_b, nullptr, tmp1, BT, PROJ_PAD, DIM);
    gemm_bf16<1><<<dim3(DIM / 128, BT / 128), 256, 0, stream>>>(
        up_b, W2T, out, nullptr, x1, nullptr, BT, DIM, PROJ_PAD);
}
